// Net_int_12421045420311
// MI455X (gfx1250) — matrix operand order repaired and checked by reading, not compiled, not run on hardware
//
#include <hip/hip_runtime.h>

// ---------------- problem constants ----------------
#define DIM      64
#define NN       16384
#define NE       32768
#define NP       65536
#define HS       132          // h1ext row stride (129 used: 128 relu vals + constant 1.0)
#define KB_TOT   258          // 8256 / 32 K-blocks  (8192 from W2 + 64 rows from b2)
#define EPS      1e-5f

typedef __attribute__((ext_vector_type(16))) __bf16 v16bf;
typedef __attribute__((ext_vector_type(8)))  __bf16 v8bf;
typedef __attribute__((ext_vector_type(8)))  float  v8f;

__device__ __forceinline__ v8f v8f_zero() {
    v8f v;
#pragma unroll
    for (int i = 0; i < 8; ++i) v[i] = 0.0f;
    return v;
}

__device__ __forceinline__ v8f wmma_bf16(v16bf a, v16bf b, v8f c) {
    // D = A(16x32 bf16) * B(32x16 bf16) + C(16x16 f32)
    return __builtin_amdgcn_wmma_f32_16x16x32_bf16(false, a, false, b, (short)0, c,
                                                   false, false);
}

__device__ __forceinline__ void atomAddF(float* p, float v) {
    __hip_atomic_fetch_add(p, v, __ATOMIC_RELAXED, __HIP_MEMORY_SCOPE_AGENT);
}

__device__ __forceinline__ float sigmf(float x) { return 1.0f / (1.0f + __expf(-x)); }

// build A fragment = stored fragment * scalar (bf16 products; compiler emits v_pk_mul_bf16)
__device__ __forceinline__ v16bf amul(v16bf f, float s) {
    v16bf a;
    __bf16 sb = (__bf16)s;
#pragma unroll
    for (int j = 0; j < 16; ++j) a[j] = (__bf16)((float)f[j] * (float)sb);
    return a;
}

// ---------------- prep kernels ----------------

__global__ void k_node_embed(const float* __restrict__ x, const float* __restrict__ Wn,
                             const float* __restrict__ bn, float* __restrict__ outf,
                             __bf16* __restrict__ outb) {
    int id = blockIdx.x * blockDim.x + threadIdx.x;
    if (id >= NN * DIM) return;
    int n = id >> 6, c = id & 63;
    float acc = bn[c];
#pragma unroll
    for (int i = 0; i < 8; ++i) acc += x[n * 8 + i] * Wn[i * DIM + c];
    acc = fmaxf(acc, 0.0f);
    outf[id] = acc;
    outb[id] = (__bf16)acc;
}

__global__ void k_deg(const int* __restrict__ ei, float* __restrict__ deg) {
    int e = blockIdx.x * blockDim.x + threadIdx.x;
    if (e >= NE) return;
    atomAddF(&deg[ei[NE + e]], 1.0f);
}

// edge MLP: ea = relu(edge_attr@We+be) [12]; h1 = relu(ea@W1+b1) [128]; h1ext[128]=1
__global__ void k_edge_mlp(const float* __restrict__ ea_in, const float* __restrict__ We,
                           const float* __restrict__ be, const float* __restrict__ W1,
                           const float* __restrict__ b1, __bf16* __restrict__ h1ext) {
    __shared__ float eas[12];
    int e = blockIdx.x;
    int t = threadIdx.x;   // 128 threads
    if (t < 12) {
        float a = be[t];
#pragma unroll
        for (int i = 0; i < 19; ++i) a += ea_in[e * 19 + i] * We[i * 12 + t];
        eas[t] = fmaxf(a, 0.0f);
    }
    __syncthreads();
    float h = b1[t];
#pragma unroll
    for (int j = 0; j < 12; ++j) h += eas[j] * W1[j * 128 + t];
    h1ext[(size_t)e * HS + t] = (__bf16)fmaxf(h, 0.0f);
    if (t == 0) h1ext[(size_t)e * HS + 128] = (__bf16)1.0f;
}

// Pack W2ext (= [W2 as 8192x64 ; b2 as 64x64]) into per-lane WMMA B-fragments:
// frag[((kb*4+nb)*32+lane)*16 + j] = B(K = kb*32 + (lane<16?0:16)+j, N = nb*16+(lane&15))
__global__ void k_pack_w2(const float* __restrict__ W2, const float* __restrict__ b2,
                          __bf16* __restrict__ frag) {
    int id = blockIdx.x * blockDim.x + threadIdx.x;
    if (id >= KB_TOT * 4 * 32 * 16) return;
    int j = id & 15, lane = (id >> 4) & 31, nb = (id >> 9) & 3, kb = id >> 11;
    int K = kb * 32 + ((lane < 16) ? 0 : 16) + j;
    int o = nb * 16 + (lane & 15);
    float v;
    if (K < 8192) {
        int k = K >> 6, i = K & 63;
        v = W2[k * 4096 + i * 64 + o];
    } else {
        v = b2[(K - 8192) * 64 + o];
    }
    frag[id] = (__bf16)v;
}

// Pack Wih^T / Whh^T ([K=64, N=192], B[K,N] = W[N*64+K]) into B-fragments.
__global__ void k_pack_gru(const float* __restrict__ Wih, const float* __restrict__ Whh,
                           __bf16* __restrict__ frag) {
    int id = blockIdx.x * blockDim.x + threadIdx.x;
    if (id >= 2 * 2 * 12 * 32 * 16) return;
    int j = id & 15, lane = (id >> 4) & 31;
    int q = id >> 9;
    int nb = q % 12, q2 = q / 12, kb = q2 & 1, mat = q2 >> 1;
    int K = kb * 32 + ((lane < 16) ? 0 : 16) + j;
    int N = nb * 16 + (lane & 15);
    const float* W = mat ? Whh : Wih;
    frag[id] = (__bf16)W[N * 64 + K];
}

// ---------------- NNConv message pass (WMMA GEMM, U synthesized on the fly) --------
// blockDim = 64 (2 waves); each wave: 32 edges x 64 outputs, K = 8256.
// K loop is unrolled by 2 so the parity into af[][] is a compile-time constant
// (dynamic indexing would demote the fragment array to scratch).
__global__ __launch_bounds__(64, 1) void k_msg(const __bf16* __restrict__ outb,
                                               const __bf16* __restrict__ h1ext,
                                               const __bf16* __restrict__ w2f,
                                               const int* __restrict__ ei,
                                               float* __restrict__ agg) {
    __shared__ __bf16 h1s[2][32][130];
    __shared__ int    dsts[2][32];

    int w     = threadIdx.x >> 5;
    int lane  = threadIdx.x & 31;
    int gwave = blockIdx.x * 2 + w;
    int ebase = gwave * 32;

    for (int e = 0; e < 32; ++e)
        for (int k = lane; k < 129; k += 32)
            h1s[w][e][k] = h1ext[(size_t)(ebase + e) * HS + k];
    if (lane < 32) dsts[w][lane] = ei[NE + ebase + lane];
    __syncthreads();

    // out[src] fragments: per M-tile t, parity p covers K%64 in {p*32..p*32+31}
    v16bf af[2][2];
#pragma unroll
    for (int t = 0; t < 2; ++t) {
        int erow = t * 16 + (lane & 15);
        int node = ei[ebase + erow];
        const __bf16* rp = outb + (size_t)node * DIM;
#pragma unroll
        for (int p = 0; p < 2; ++p) {
            int cb = p * 32 + ((lane < 16) ? 0 : 16);
            v8bf lo = *(const v8bf*)(rp + cb);
            v8bf hi = *(const v8bf*)(rp + cb + 8);
            af[t][p] = __builtin_shufflevector(lo, hi, 0, 1, 2, 3, 4, 5, 6, 7,
                                               8, 9, 10, 11, 12, 13, 14, 15);
        }
    }

    v8f acc[2][4];
#pragma unroll
    for (int t = 0; t < 2; ++t)
#pragma unroll
        for (int n = 0; n < 4; ++n) acc[t][n] = v8f_zero();

    // 129 iterations; iteration k covers K-blocks kb=2k (parity 0) and kb=2k+1 (parity 1)
    for (int k = 0; k < 129; ++k) {
        const __bf16* bp = w2f + (size_t)k * 4096 + lane * 16;   // kb0 fragments
        {
            int kpf = (k + 8 < 129) ? k + 8 : 128;
            __builtin_prefetch(w2f + (size_t)kpf * 4096 + lane * 16, 0, 1);
        }
        v16bf b00 = *(const v16bf*)(bp + 0 * 512);
        v16bf b01 = *(const v16bf*)(bp + 1 * 512);
        v16bf b02 = *(const v16bf*)(bp + 2 * 512);
        v16bf b03 = *(const v16bf*)(bp + 3 * 512);
        v16bf b10 = *(const v16bf*)(bp + 4 * 512);               // kb1 fragments
        v16bf b11 = *(const v16bf*)(bp + 5 * 512);
        v16bf b12 = *(const v16bf*)(bp + 6 * 512);
        v16bf b13 = *(const v16bf*)(bp + 7 * 512);

        float s0 = (float)h1s[w][lane & 15][k];          // M-tile 0 row scalar
        float s1 = (float)h1s[w][16 + (lane & 15)][k];   // M-tile 1 row scalar

        // tile 0
        {
            v16bf a0 = amul(af[0][0], s0);
            acc[0][0] = wmma_bf16(a0, b00, acc[0][0]);
            acc[0][1] = wmma_bf16(a0, b01, acc[0][1]);
            acc[0][2] = wmma_bf16(a0, b02, acc[0][2]);
            acc[0][3] = wmma_bf16(a0, b03, acc[0][3]);
            v16bf a1 = amul(af[0][1], s0);
            acc[0][0] = wmma_bf16(a1, b10, acc[0][0]);
            acc[0][1] = wmma_bf16(a1, b11, acc[0][1]);
            acc[0][2] = wmma_bf16(a1, b12, acc[0][2]);
            acc[0][3] = wmma_bf16(a1, b13, acc[0][3]);
        }
        // tile 1
        {
            v16bf a0 = amul(af[1][0], s1);
            acc[1][0] = wmma_bf16(a0, b00, acc[1][0]);
            acc[1][1] = wmma_bf16(a0, b01, acc[1][1]);
            acc[1][2] = wmma_bf16(a0, b02, acc[1][2]);
            acc[1][3] = wmma_bf16(a0, b03, acc[1][3]);
            v16bf a1 = amul(af[1][1], s1);
            acc[1][0] = wmma_bf16(a1, b10, acc[1][0]);
            acc[1][1] = wmma_bf16(a1, b11, acc[1][1]);
            acc[1][2] = wmma_bf16(a1, b12, acc[1][2]);
            acc[1][3] = wmma_bf16(a1, b13, acc[1][3]);
        }
    }

    // scatter-add messages: C-tile element (VGPR v, lane) = (row M, col N)
#pragma unroll
    for (int t = 0; t < 2; ++t)
#pragma unroll
        for (int v = 0; v < 8; ++v) {
            int row = t * 16 + ((lane < 16) ? v : v + 8);
            int d   = dsts[w][row];
#pragma unroll
            for (int n = 0; n < 4; ++n) {
                int col = n * 16 + (lane & 15);
                atomAddF(&agg[(size_t)d * DIM + col], acc[t][n][v]);
            }
        }
}

// ---------------- fused m = relu(agg/deg + b) + GRU cell (WMMA) -------------------
__global__ __launch_bounds__(64, 1) void k_gru(const float* __restrict__ agg,
                                               const float* __restrict__ deg,
                                               const float* __restrict__ conv_b,
                                               const __bf16* __restrict__ gf,
                                               const float* __restrict__ bih,
                                               const float* __restrict__ bhh,
                                               float* __restrict__ outf,
                                               __bf16* __restrict__ outb) {
    int w     = threadIdx.x >> 5;
    int lane  = threadIdx.x & 31;
    int nbase = (blockIdx.x * 2 + w) * 16;

    int   node = nbase + (lane & 15);
    float inv  = 1.0f / fmaxf(deg[node], 1.0f);

    v16bf am[2], ah[2];
#pragma unroll
    for (int p = 0; p < 2; ++p) {
        int cb = p * 32 + ((lane < 16) ? 0 : 16);
#pragma unroll
        for (int j = 0; j < 16; ++j) {
            int c   = cb + j;
            float m = fmaxf(agg[(size_t)node * DIM + c] * inv + conv_b[c], 0.0f);
            am[p][j] = (__bf16)m;
            ah[p][j] = (__bf16)outf[(size_t)node * DIM + c];
        }
    }

    v8f aRZ[8], aXN[4], aHN[4];
#pragma unroll
    for (int i = 0; i < 8; ++i) aRZ[i] = v8f_zero();
#pragma unroll
    for (int i = 0; i < 4; ++i) { aXN[i] = v8f_zero(); aHN[i] = v8f_zero(); }

#pragma unroll
    for (int kb = 0; kb < 2; ++kb)
#pragma unroll
        for (int nb = 0; nb < 12; ++nb) {
            v16bf bw = *(const v16bf*)(gf + ((size_t)((0 * 2 + kb) * 12 + nb) * 32 + lane) * 16);
            v16bf bh = *(const v16bf*)(gf + ((size_t)((1 * 2 + kb) * 12 + nb) * 32 + lane) * 16);
            if (nb < 8) {   // r,z gates: only gx+gh is used -> shared accumulator
                aRZ[nb] = wmma_bf16(am[kb], bw, aRZ[nb]);
                aRZ[nb] = wmma_bf16(ah[kb], bh, aRZ[nb]);
            } else {        // n gate: keep gx_n, gh_n separate (r * gh_n)
                aXN[nb - 8] = wmma_bf16(am[kb], bw, aXN[nb - 8]);
                aHN[nb - 8] = wmma_bf16(ah[kb], bh, aHN[nb - 8]);
            }
        }

#pragma unroll
    for (int j = 0; j < 4; ++j) {
        int   c  = j * 16 + (lane & 15);                // gate-local column
        float br = bih[c] + bhh[c];
        float bz = bih[64 + c] + bhh[64 + c];
        float bx = bih[128 + c], bg = bhh[128 + c];
#pragma unroll
        for (int v = 0; v < 8; ++v) {
            int   row  = nbase + ((lane < 16) ? v : v + 8);
            float r    = sigmf(aRZ[j][v] + br);
            float z    = sigmf(aRZ[j + 4][v] + bz);
            float n    = tanhf(aXN[j][v] + bx + r * (aHN[j][v] + bg));
            float hold = outf[(size_t)row * DIM + c];
            float hnew = (1.0f - z) * n + z * hold;
            outf[(size_t)row * DIM + c] = hnew;
            outb[(size_t)row * DIM + c] = (__bf16)hnew;
        }
    }
}

// ---------------- readout ----------------
__global__ void k_r1(const float* __restrict__ outf, const int* __restrict__ ei3,
                     float* __restrict__ stats) {
    int c  = threadIdx.x;            // 0..191
    int tc = c & 63;
    int pb = blockIdx.x * 256;
    float s = 0.0f, sq = 0.0f;
    for (int p = 0; p < 256; ++p) {
        int i0 = ei3[pb + p], i1 = ei3[NP + pb + p];
        float a = outf[(size_t)i0 * DIM + tc];
        float b = outf[(size_t)i1 * DIM + tc];
        float y = (c < 64) ? 0.5f * (a + b) : ((c < 128) ? a * b : (a - b) * (a - b));
        s += y; sq += y * y;
    }
    atomAddF(&stats[c], s);
    atomAddF(&stats[192 + c], sq);
}

__global__ void k_r2(const float* __restrict__ outf, const int* __restrict__ ei3,
                     const float* __restrict__ ea3, const float* __restrict__ Wlw,
                     const float* __restrict__ Wlb, const float* __restrict__ gamma,
                     const float* __restrict__ beta, const float* __restrict__ stats,
                     float* __restrict__ dout) {
    __shared__ float smu[192], sg[192], sb[192], sW[8 * 192];
    int tid = threadIdx.x;
    for (int i = tid; i < 192; i += 256) {
        float mu  = stats[i] * (1.0f / NP);
        float var = stats[192 + i] * (1.0f / NP) - mu * mu;
        smu[i] = mu;
        sg[i]  = rsqrtf(var + EPS) * gamma[i];
        sb[i]  = beta[i];
    }
    for (int i = tid; i < 8 * 192; i += 256) sW[i] = Wlw[i];
    __syncthreads();

    int p  = blockIdx.x * 256 + tid;
    int i0 = ei3[p], i1 = ei3[NP + p];
    float e3[8];
#pragma unroll
    for (int i = 0; i < 8; ++i) e3[i] = ea3[p * 8 + i];

    float acc = 0.0f;
    for (int tc = 0; tc < 64; ++tc) {
        float a = outf[(size_t)i0 * DIM + tc];
        float b = outf[(size_t)i1 * DIM + tc];
#pragma unroll
        for (int g = 0; g < 3; ++g) {
            int   c = g * 64 + tc;
            float y = (g == 0) ? 0.5f * (a + b) : ((g == 1) ? a * b : (a - b) * (a - b));
            float yn = (y - smu[c]) * sg[c] + sb[c];
            float wv = 0.0f;
#pragma unroll
            for (int i = 0; i < 8; ++i) wv += e3[i] * sW[i * 192 + c];
            acc += yn * wv;
        }
    }
    float bias = 0.0f;
#pragma unroll
    for (int i = 0; i < 8; ++i) bias += e3[i] * Wlb[i];
    dout[p] = acc + bias;
}

// ---------------- launch ----------------
static size_t alignup(size_t x) { return (x + 255) & ~(size_t)255; }

extern "C" void kernel_launch(void* const* d_in, const int* in_sizes, int n_in,
                              void* d_out, int out_size, void* d_ws, size_t ws_size,
                              hipStream_t stream) {
    const float* x      = (const float*)d_in[0];
    const float* eattr  = (const float*)d_in[1];
    const float* ea3    = (const float*)d_in[2];
    const int*   ei     = (const int*)d_in[3];
    const int*   ei3    = (const int*)d_in[4];
    const float* Wn     = (const float*)d_in[5];
    const float* bn     = (const float*)d_in[6];
    const float* We     = (const float*)d_in[7];
    const float* be     = (const float*)d_in[8];
    const float* W1     = (const float*)d_in[9];
    const float* b1     = (const float*)d_in[10];
    const float* W2     = (const float*)d_in[11];
    const float* b2     = (const float*)d_in[12];
    const float* conv_b = (const float*)d_in[13];
    const float* Wih    = (const float*)d_in[14];
    const float* Whh    = (const float*)d_in[15];
    const float* bih    = (const float*)d_in[16];
    const float* bhh    = (const float*)d_in[17];
    const float* Wlw    = (const float*)d_in[18];
    const float* Wlb    = (const float*)d_in[19];
    const float* gamma  = (const float*)d_in[20];
    const float* beta   = (const float*)d_in[21];
    float* dout = (float*)d_out;

    char*  ws  = (char*)d_ws;
    size_t cur = 0;
    float*  outf  = (float*)(ws + cur);  cur = alignup(cur + (size_t)NN * DIM * 4);
    float*  agg   = (float*)(ws + cur);  cur = alignup(cur + (size_t)NN * DIM * 4);
    __bf16* outb  = (__bf16*)(ws + cur); cur = alignup(cur + (size_t)NN * DIM * 2);
    __bf16* h1ext = (__bf16*)(ws + cur); cur = alignup(cur + (size_t)NE * HS * 2);
    __bf16* w2f   = (__bf16*)(ws + cur); cur = alignup(cur + (size_t)KB_TOT * 4 * 32 * 16 * 2);
    __bf16* gf    = (__bf16*)(ws + cur); cur = alignup(cur + (size_t)2 * 2 * 12 * 32 * 16 * 2);
    float*  deg   = (float*)(ws + cur);  cur = alignup(cur + (size_t)NN * 4);
    float*  stats = (float*)(ws + cur);  cur = alignup(cur + 384 * 4);

    // prep
    hipMemsetAsync(deg, 0, (size_t)NN * 4, stream);
    hipMemsetAsync(stats, 0, 384 * 4, stream);
    k_node_embed<<<(NN * DIM + 255) / 256, 256, 0, stream>>>(x, Wn, bn, outf, outb);
    k_deg<<<(NE + 255) / 256, 256, 0, stream>>>(ei, deg);
    k_edge_mlp<<<NE, 128, 0, stream>>>(eattr, We, be, W1, b1, h1ext);
    k_pack_w2<<<(KB_TOT * 4 * 32 * 16 + 255) / 256, 256, 0, stream>>>(W2, b2, w2f);
    k_pack_gru<<<(2 * 2 * 12 * 32 * 16 + 255) / 256, 256, 0, stream>>>(Wih, Whh, gf);

    // 3 message-passing + GRU iterations
    for (int it = 0; it < 3; ++it) {
        hipMemsetAsync(agg, 0, (size_t)NN * DIM * 4, stream);
        k_msg<<<NE / 64, 64, 0, stream>>>(outb, h1ext, w2f, ei, agg);
        k_gru<<<NN / 32, 64, 0, stream>>>(agg, deg, conv_b, gf, bih, bhh, outf, outb);
    }

    // readout
    k_r1<<<NP / 256, 192, 0, stream>>>(outf, ei3, stats);
    k_r2<<<NP / 256, 256, 0, stream>>>(outf, ei3, ea3, Wlw, Wlb, gamma, beta, stats, dout);
}
